// GraphMixerEncoder_16836271801000
// MI455X (gfx1250) — matrix operand order repaired and checked by reading, not compiled, not run on hardware
//
#include <hip/hip_runtime.h>
#include <math.h>

#define NNODES 100000
#define NEDGES 1600000

typedef __attribute__((ext_vector_type(16))) __bf16 v16bf;
typedef __attribute__((ext_vector_type(8)))  float  v8f;
typedef __attribute__((ext_vector_type(4)))  unsigned v4u;

// ---------- helpers ----------

__device__ __forceinline__ unsigned short f2bf(float f) {
  unsigned u = __float_as_uint(f);
  u += 0x7FFFu + ((u >> 16) & 1u);        // round-to-nearest-even
  return (unsigned short)(u >> 16);
}

__device__ __forceinline__ v8f zero8() {
  v8f z;
#pragma unroll
  for (int i = 0; i < 8; ++i) z[i] = 0.f;
  return z;
}

__device__ __forceinline__ v8f wmma_bf16(v16bf a, v16bf b, v8f c) {
  return __builtin_amdgcn_wmma_f32_16x16x32_bf16(false, a, false, b, (short)0, c,
                                                 false, false);
}

// A fragment (16x32 bf16). rowbase = start of this lane's row (m = lane&15) in a
// row-major [16][Kpad] bf16 buffer (LDS or global), 16B aligned. ISA layout:
//   lane l, vgpr v -> K = (v%4)*2 + (l/16)*8 + (v/4)*16  (pairs of K packed per dword)
// dwords 0..3 contiguous, dwords 8..11 contiguous -> two b128 loads.
__device__ __forceinline__ v16bf load_a_frag(const unsigned short* rowbase, int ktile,
                                             int lane) {
  const int hi = lane >> 4;
  const v4u* p = (const v4u*)(rowbase + ktile * 32 + hi * 8);
  union { v16bf v; v4u q[2]; } r;
  r.q[0] = p[0];          // dwords 0..3   (K = 0..7   within half)
  r.q[1] = p[2];          // dwords 8..11  (K = 16..23 within half)
  return r.v;
}

// B fragment (32x16 bf16) from transposed weight buffer Wt[Npad][Kpad] bf16.
//   lane l (n = l&15, hi = l>>4), vgpr v -> K = (v%4)*2 + hi*8 + (v/4)*16, N = ntile*16 + n
// dwords 0..3 and 8..11 -> two b128 loads.
__device__ __forceinline__ v16bf load_b_frag(const unsigned short* wt, int kpad,
                                             int ntile, int ktile, int lane) {
  const int n = lane & 15, hi = lane >> 4;
  const v4u* p = (const v4u*)(wt + (ntile * 16 + n) * kpad + ktile * 32 + hi * 8);
  union { v16bf v; v4u q[2]; } r;
  r.q[0] = p[0];
  r.q[1] = p[2];
  return r.v;
}

// sum across a 16-lane half (matches C/D layout: row M = r + 8*(lane>>4))
__device__ __forceinline__ float half16_sum(float x) {
  x += __shfl_xor(x, 1, 32);
  x += __shfl_xor(x, 2, 32);
  x += __shfl_xor(x, 4, 32);
  x += __shfl_xor(x, 8, 32);
  return x;
}

__device__ __forceinline__ float wave_sum(float x) {
  x += __shfl_xor(x, 1, 32);
  x += __shfl_xor(x, 2, 32);
  x += __shfl_xor(x, 4, 32);
  x += __shfl_xor(x, 8, 32);
  x += __shfl_xor(x, 16, 32);
  return x;
}

__device__ __forceinline__ float gelu_exact(float x) {
  return 0.5f * x * (1.0f + erff(x * 0.70710678118654752f));
}

__device__ __forceinline__ int imin(int a, int b) { return a < b ? a : b; }

// Stage W[K][Nn] (row-major, f32) into LDS as transposed bf16 Wt[Npad][Kpad],
// zero padded.
__device__ void stage_wt(unsigned short* dst, const float* W, int K, int Nn, int Kpad,
                         int Npad, int tid, int nthr) {
  const int total = Npad * Kpad;
  for (int i = tid; i < total; i += nthr) {
    const int n = i / Kpad, k = i % Kpad;
    const float v = (n < Nn && k < K) ? W[k * Nn + n] : 0.0f;
    dst[i] = f2bf(v);
  }
}

// ---------- kernel 0: zero scatter buffers ----------

__global__ void zero_kernel(float* p, size_t n) {
  size_t i = (size_t)blockIdx.x * blockDim.x + threadIdx.x;
  const size_t stride = (size_t)gridDim.x * blockDim.x;
  for (; i < n; i += stride) p[i] = 0.f;
}

// ---------- kernel 1: edge token mixer + scatter ----------
// 4 waves/block, 16 edges/wave. K1 pads: tokens K=224, hidden K=128, ffn-mid K=64.
#define EW 4
#define EDGE_LDS ((112*224 + 64*128 + 112*64 + EW*16*224 + EW*16*128 + EW*16*64)*2 + EW*16*8)

__global__ void edge_kernel(const float* __restrict__ lu, const int* __restrict__ ei,
                            const float* __restrict__ t, const float* __restrict__ msg,
                            const float* __restrict__ tw, const float* __restrict__ tb,
                            const float* __restrict__ featW, const float* __restrict__ featb,
                            const float* __restrict__ tlng, const float* __restrict__ tlnb,
                            const float* __restrict__ tf0W, const float* __restrict__ tf0b,
                            const float* __restrict__ tf1W, const float* __restrict__ tf1b,
                            float* __restrict__ h_sum, float* __restrict__ cnt) {
  extern __shared__ unsigned short smem[];
  unsigned short* featWt = smem;                       // [112][224]
  unsigned short* tf0t   = featWt + 112 * 224;         // [64][128]
  unsigned short* tf1t   = tf0t + 64 * 128;            // [112][64]
  unsigned short* tok    = tf1t + 112 * 64;            // EW*[16][224]
  unsigned short* hbuf   = tok + EW * 16 * 224;        // EW*[16][128]
  unsigned short* gbuf   = hbuf + EW * 16 * 128;       // EW*[16][64]
  float* relbuf = (float*)(gbuf + EW * 16 * 64);       // EW*16
  int*   dstbuf = (int*)(relbuf + EW * 16);            // EW*16

  const int tid = threadIdx.x, nthr = blockDim.x;
  const int w = tid >> 5, lane = tid & 31;
  const int n16 = lane & 15, hi = lane >> 4;
  const int blockBase = blockIdx.x * (EW * 16);
  const int base = blockBase + w * 16;

  // prefetch the msg tile into cache while weights stage (global_prefetch_b8)
  {
    const char* mb = (const char*)(msg + (size_t)blockBase * 100);
    const size_t bytes = (size_t)(EW * 16) * 100 * 4;
    for (size_t off = (size_t)tid * 64; off < bytes; off += (size_t)nthr * 64)
      __builtin_prefetch(mb + off, 0, 1);
  }

  stage_wt(featWt, featW, 200, 100, 224, 112, tid, nthr);
  stage_wt(tf0t, tf0W, 100, 50, 128, 64, tid, nthr);
  stage_wt(tf1t, tf1W, 50, 100, 64, 112, tid, nthr);

  if (lane < 16) {
    const int e = base + lane;
    const int s = ei[e];
    relbuf[w * 16 + lane] = lu[s] - t[e];
    dstbuf[w * 16 + lane] = ei[NEDGES + e];
  }
  __syncthreads();

  // build edge tokens: [te(100) | msg(100) | 0 pad] -> bf16 [16][224]
  unsigned short* mytok = tok + w * 16 * 224;
#pragma unroll 1
  for (int m = 0; m < 16; ++m) {
    const float rel = relbuf[w * 16 + m];
    const float* mrow = msg + (size_t)(base + m) * 100;
    for (int j = lane; j < 224; j += 32) {
      float v;
      if (j < 100)       v = cosf(rel * tw[j] + tb[j]);
      else if (j < 200)  v = mrow[j - 100];
      else               v = 0.f;
      mytok[m * 224 + j] = f2bf(v);
    }
  }
  __syncthreads();

  // GEMM1: h = tokens @ feat_W  (16 x 224 x 112)
  v8f acc[7];
#pragma unroll
  for (int i = 0; i < 7; ++i) acc[i] = zero8();
  {
    const unsigned short* myrow = mytok + n16 * 224;
#pragma unroll 1
    for (int kt = 0; kt < 7; ++kt) {
      const v16bf a = load_a_frag(myrow, kt, lane);
#pragma unroll
      for (int nt = 0; nt < 7; ++nt)
        acc[nt] = wmma_bf16(a, load_b_frag(featWt, 224, nt, kt, lane), acc[nt]);
    }
  }

  // bias + token LayerNorm (per row, 100 valid channels)
  float mu[8], rstd[8];
#pragma unroll
  for (int r = 0; r < 8; ++r) {
    float s1 = 0.f, s2 = 0.f;
#pragma unroll
    for (int nt = 0; nt < 7; ++nt) {
      const int Nc = nt * 16 + n16;
      const float xv = (Nc < 100) ? (acc[nt][r] + featb[Nc]) : 0.f;
      s1 += xv; s2 += xv * xv;
    }
    s1 = half16_sum(s1); s2 = half16_sum(s2);
    const float m_ = s1 * 0.01f;
    mu[r] = m_;
    rstd[r] = rsqrtf(s2 * 0.01f - m_ * m_ + 1e-5f);
  }

  unsigned short* whb = hbuf + w * 16 * 128;
#pragma unroll
  for (int nt = 0; nt < 7; ++nt) {
    const int Nc = nt * 16 + n16;
#pragma unroll
    for (int r = 0; r < 8; ++r) {
      const int M = r + 8 * hi;
      float hn = 0.f;
      if (Nc < 100) {
        const float xv = acc[nt][r] + featb[Nc];
        hn = (xv - mu[r]) * rstd[r] * tlng[Nc] + tlnb[Nc];
      }
      acc[nt][r] = hn;                       // keep residual (post-LN h)
      whb[M * 128 + Nc] = f2bf(hn);
    }
  }
  for (int idx = lane; idx < 256; idx += 32) {   // zero cols 112..127
    const int M = idx >> 4, c = 112 + (idx & 15);
    whb[M * 128 + c] = 0;
  }
  __syncthreads();

  // GEMM2: gelu(h @ tf0_W + b)  (16 x 128 x 64)
  v8f acc2[4];
#pragma unroll
  for (int i = 0; i < 4; ++i) acc2[i] = zero8();
  {
    const unsigned short* myrow = whb + n16 * 128;
#pragma unroll 1
    for (int kt = 0; kt < 4; ++kt) {
      const v16bf a = load_a_frag(myrow, kt, lane);
#pragma unroll
      for (int nt = 0; nt < 4; ++nt)
        acc2[nt] = wmma_bf16(a, load_b_frag(tf0t, 128, nt, kt, lane), acc2[nt]);
    }
  }
  unsigned short* wgb = gbuf + w * 16 * 64;
#pragma unroll
  for (int nt = 0; nt < 4; ++nt) {
    const int Nc = nt * 16 + n16;
#pragma unroll
    for (int r = 0; r < 8; ++r) {
      const int M = r + 8 * hi;
      const float g = (Nc < 50) ? gelu_exact(acc2[nt][r] + tf0b[Nc]) : 0.f;
      wgb[M * 64 + Nc] = f2bf(g);
    }
  }
  __syncthreads();

  // GEMM3: g @ tf1_W  (16 x 64 x 112), add residual + bias, scatter-add
  v8f acc3[7];
#pragma unroll
  for (int i = 0; i < 7; ++i) acc3[i] = zero8();
  {
    const unsigned short* myrow = wgb + n16 * 64;
#pragma unroll 1
    for (int kt = 0; kt < 2; ++kt) {
      const v16bf a = load_a_frag(myrow, kt, lane);
#pragma unroll
      for (int nt = 0; nt < 7; ++nt)
        acc3[nt] = wmma_bf16(a, load_b_frag(tf1t, 64, nt, kt, lane), acc3[nt]);
    }
  }

  int drow[8];
#pragma unroll
  for (int r = 0; r < 8; ++r) drow[r] = dstbuf[w * 16 + r + 8 * hi];

#pragma unroll
  for (int nt = 0; nt < 7; ++nt) {
    const int Nc = nt * 16 + n16;
    if (Nc < 100) {
#pragma unroll
      for (int r = 0; r < 8; ++r) {
        const float hv = acc[nt][r] + acc3[nt][r] + tf1b[Nc];
        atomicAdd(&h_sum[(size_t)drow[r] * 100 + Nc], hv);
      }
    }
  }
  if (lane < 16) atomicAdd(&cnt[dstbuf[w * 16 + lane]], 1.0f);
}

// ---------- kernel 2: node mean + channel-LN + cf0 GEMM + GELU -> hc1 (bf16) ----------
#define K2A_LDS ((416*128 + EW*16*128) * 2)

__global__ void node_ffn0_kernel(const float* __restrict__ h_sum,
                                 const float* __restrict__ cnt,
                                 const float* __restrict__ clng,
                                 const float* __restrict__ clnb,
                                 const float* __restrict__ cf0W,
                                 const float* __restrict__ cf0b,
                                 unsigned short* __restrict__ hc1) {
  extern __shared__ unsigned short smem[];
  unsigned short* cf0t = smem;                 // [416][128]
  unsigned short* hln  = cf0t + 416 * 128;     // EW*[16][128]

  const int tid = threadIdx.x, nthr = blockDim.x;
  const int w = tid >> 5, lane = tid & 31;
  const int n16 = lane & 15, hi = lane >> 4;
  const int base = blockIdx.x * (EW * 16) + w * 16;

  stage_wt(cf0t, cf0W, 100, 400, 128, 416, tid, nthr);

  unsigned short* whl = hln + w * 16 * 128;
#pragma unroll 1
  for (int m = 0; m < 16; ++m) {
    const int n = imin(base + m, NNODES - 1);
    const float inv = 1.0f / fmaxf(cnt[n], 1.0f);
    float v[4];
    float s1 = 0.f, s2 = 0.f;
#pragma unroll
    for (int q = 0; q < 4; ++q) {
      const int j = lane + q * 32;
      const float xv = (j < 100) ? h_sum[(size_t)n * 100 + j] * inv : 0.f;
      v[q] = xv; s1 += xv; s2 += xv * xv;
    }
    s1 = wave_sum(s1); s2 = wave_sum(s2);
    const float mu = s1 * 0.01f;
    const float rs = rsqrtf(s2 * 0.01f - mu * mu + 1e-5f);
#pragma unroll
    for (int q = 0; q < 4; ++q) {
      const int j = lane + q * 32;
      const float hn = (j < 100) ? ((v[q] - mu) * rs * clng[j] + clnb[j]) : 0.f;
      whl[m * 128 + j] = f2bf(hn);
    }
  }
  __syncthreads();

  v16bf a[4];
  {
    const unsigned short* myrow = whl + n16 * 128;
#pragma unroll
    for (int kt = 0; kt < 4; ++kt) a[kt] = load_a_frag(myrow, kt, lane);
  }

#pragma unroll 1
  for (int nt = 0; nt < 26; ++nt) {            // N = 416 (400 valid)
    v8f acc = zero8();
#pragma unroll
    for (int kt = 0; kt < 4; ++kt)
      acc = wmma_bf16(a[kt], load_b_frag(cf0t, 128, nt, kt, lane), acc);
    const int Nc = nt * 16 + n16;
    const float b = (Nc < 400) ? cf0b[Nc] : 0.f;
#pragma unroll
    for (int r = 0; r < 8; ++r) {
      const int n = imin(base + r + 8 * hi, NNODES - 1);
      const float g = (Nc < 400) ? gelu_exact(acc[r] + b) : 0.f;
      hc1[(size_t)n * 416 + Nc] = f2bf(g);
    }
  }
}

// ---------- kernel 3: cf1 GEMM + residual + proj GEMM -> out_pre ----------
#define K2B_LDS ((112*416 + 112*128 + EW*16*128) * 2)

__global__ void node_ffn1_kernel(const float* __restrict__ h_sum,
                                 const float* __restrict__ cnt,
                                 const unsigned short* __restrict__ hc1,
                                 const float* __restrict__ cf1W,
                                 const float* __restrict__ cf1b,
                                 const float* __restrict__ projW,
                                 const float* __restrict__ projb,
                                 float* __restrict__ out_pre) {
  extern __shared__ unsigned short smem[];
  unsigned short* cf1t  = smem;                 // [112][416]
  unsigned short* projt = cf1t + 112 * 416;     // [112][128]
  unsigned short* h2buf = projt + 112 * 128;    // EW*[16][128]

  const int tid = threadIdx.x, nthr = blockDim.x;
  const int w = tid >> 5, lane = tid & 31;
  const int n16 = lane & 15, hi = lane >> 4;
  const int base = blockIdx.x * (EW * 16) + w * 16;

  stage_wt(cf1t, cf1W, 400, 100, 416, 112, tid, nthr);
  stage_wt(projt, projW, 100, 100, 128, 112, tid, nthr);
  for (int i = tid; i < EW * 16 * 128; i += nthr) h2buf[i] = 0;
  __syncthreads();

  // hc = hc1 @ cf1_W  (16 x 416 x 112), A read straight from global bf16 ws
  v8f acc[7];
#pragma unroll
  for (int i = 0; i < 7; ++i) acc[i] = zero8();
  {
    const int rowNode = imin(base + n16, NNODES - 1);
    const unsigned short* myrow = hc1 + (size_t)rowNode * 416;
#pragma unroll 1
    for (int kt = 0; kt < 13; ++kt) {
      const v16bf a = load_a_frag(myrow, kt, lane);
#pragma unroll
      for (int nt = 0; nt < 7; ++nt)
        acc[nt] = wmma_bf16(a, load_b_frag(cf1t, 416, nt, kt, lane), acc[nt]);
    }
  }

  // h2 = h_node + hc + b ; stage bf16 for proj GEMM
  unsigned short* wh2 = h2buf + w * 16 * 128;
#pragma unroll
  for (int nt = 0; nt < 7; ++nt) {
    const int Nc = nt * 16 + n16;
    if (Nc < 100) {
#pragma unroll
      for (int r = 0; r < 8; ++r) {
        const int M = r + 8 * hi;
        const int n = imin(base + M, NNODES - 1);
        const float hn = h_sum[(size_t)n * 100 + Nc] / fmaxf(cnt[n], 1.0f);
        wh2[M * 128 + Nc] = f2bf(hn + acc[nt][r] + cf1b[Nc]);
      }
    }
  }
  __syncthreads();

  // out_pre = h2 @ proj_W + proj_b  (16 x 128 x 112)
  v8f accp[7];
#pragma unroll
  for (int i = 0; i < 7; ++i) accp[i] = zero8();
  {
    const unsigned short* myrow = wh2 + n16 * 128;
#pragma unroll 1
    for (int kt = 0; kt < 4; ++kt) {
      const v16bf a = load_a_frag(myrow, kt, lane);
#pragma unroll
      for (int nt = 0; nt < 7; ++nt)
        accp[nt] = wmma_bf16(a, load_b_frag(projt, 128, nt, kt, lane), accp[nt]);
    }
  }
#pragma unroll
  for (int nt = 0; nt < 7; ++nt) {
    const int Nc = nt * 16 + n16;
    if (Nc < 100) {
#pragma unroll
      for (int r = 0; r < 8; ++r) {
        const int nn = base + r + 8 * hi;
        if (nn < NNODES) out_pre[(size_t)nn * 100 + Nc] = accp[nt][r] + projb[Nc];
      }
    }
  }
}

// ---------- kernel 4: out = LN(out_pre) + x @ res_W + res_b ----------
#define K2C_LDS ((112*128 + EW*16*128) * 2)

__global__ void node_out_kernel(const float* __restrict__ x,
                                const float* __restrict__ out_pre,
                                const float* __restrict__ resW,
                                const float* __restrict__ resb,
                                const float* __restrict__ olng,
                                const float* __restrict__ olnb,
                                float* __restrict__ out) {
  extern __shared__ unsigned short smem[];
  unsigned short* rest = smem;             // [112][128]
  unsigned short* xb   = rest + 112 * 128; // EW*[16][128]

  const int tid = threadIdx.x, nthr = blockDim.x;
  const int w = tid >> 5, lane = tid & 31;
  const int n16 = lane & 15, hi = lane >> 4;
  const int base = blockIdx.x * (EW * 16) + w * 16;

  stage_wt(rest, resW, 100, 100, 128, 112, tid, nthr);

  unsigned short* wxb = xb + w * 16 * 128;
#pragma unroll 1
  for (int m = 0; m < 16; ++m) {
    const int n = imin(base + m, NNODES - 1);
    for (int j = lane; j < 128; j += 32)
      wxb[m * 128 + j] = f2bf(j < 100 ? x[(size_t)n * 100 + j] : 0.f);
  }
  __syncthreads();

  // res GEMM: x @ res_W  (16 x 128 x 112)
  v8f accr[7];
#pragma unroll
  for (int i = 0; i < 7; ++i) accr[i] = zero8();
  {
    const unsigned short* myrow = wxb + n16 * 128;
#pragma unroll 1
    for (int kt = 0; kt < 4; ++kt) {
      const v16bf a = load_a_frag(myrow, kt, lane);
#pragma unroll
      for (int nt = 0; nt < 7; ++nt)
        accr[nt] = wmma_bf16(a, load_b_frag(rest, 128, nt, kt, lane), accr[nt]);
    }
  }

  // gather out_pre in C-layout, LayerNorm per row, combine, store
  float val[7][8];
  float mu[8], rs[8];
#pragma unroll
  for (int r = 0; r < 8; ++r) {
    const int n = imin(base + r + 8 * hi, NNODES - 1);
    float s1 = 0.f, s2 = 0.f;
#pragma unroll
    for (int nt = 0; nt < 7; ++nt) {
      const int Nc = nt * 16 + n16;
      const float v = (Nc < 100) ? out_pre[(size_t)n * 100 + Nc] : 0.f;
      val[nt][r] = v; s1 += v; s2 += v * v;
    }
    s1 = half16_sum(s1); s2 = half16_sum(s2);
    const float m_ = s1 * 0.01f;
    mu[r] = m_;
    rs[r] = rsqrtf(s2 * 0.01f - m_ * m_ + 1e-5f);
  }
#pragma unroll
  for (int nt = 0; nt < 7; ++nt) {
    const int Nc = nt * 16 + n16;
    if (Nc < 100) {
#pragma unroll
      for (int r = 0; r < 8; ++r) {
        const int nn = base + r + 8 * hi;
        if (nn < NNODES)
          out[(size_t)nn * 100 + Nc] =
              (val[nt][r] - mu[r]) * rs[r] * olng[Nc] + olnb[Nc] + accr[nt][r] + resb[Nc];
      }
    }
  }
}

// ---------- launch ----------

extern "C" void kernel_launch(void* const* d_in, const int* in_sizes, int n_in,
                              void* d_out, int out_size, void* d_ws, size_t ws_size,
                              hipStream_t stream) {
  (void)in_sizes; (void)n_in; (void)out_size; (void)ws_size;
  const float* x     = (const float*)d_in[0];
  const float* lu    = (const float*)d_in[1];
  const int*   ei    = (const int*)d_in[2];
  const float* t     = (const float*)d_in[3];
  const float* msg   = (const float*)d_in[4];
  const float* tw    = (const float*)d_in[5];
  const float* tb    = (const float*)d_in[6];
  const float* featW = (const float*)d_in[7];
  const float* featb = (const float*)d_in[8];
  const float* tlng  = (const float*)d_in[9];
  const float* tlnb  = (const float*)d_in[10];
  const float* tf0W  = (const float*)d_in[11];
  const float* tf0b  = (const float*)d_in[12];
  const float* tf1W  = (const float*)d_in[13];
  const float* tf1b  = (const float*)d_in[14];
  const float* clng  = (const float*)d_in[15];
  const float* clnb  = (const float*)d_in[16];
  const float* cf0W  = (const float*)d_in[17];
  const float* cf0b  = (const float*)d_in[18];
  const float* cf1W  = (const float*)d_in[19];
  const float* cf1b  = (const float*)d_in[20];
  const float* projW = (const float*)d_in[21];
  const float* projb = (const float*)d_in[22];
  const float* resW  = (const float*)d_in[23];
  const float* resb  = (const float*)d_in[24];
  const float* olng  = (const float*)d_in[25];
  const float* olnb  = (const float*)d_in[26];

  // workspace layout
  float* h_sum = (float*)d_ws;                                  // N*100 f32
  float* cnt   = h_sum + (size_t)NNODES * 100;                  // N f32
  unsigned short* hc1 = (unsigned short*)(cnt + NNODES);        // N*416 bf16
  float* out_pre = (float*)(hc1 + (size_t)NNODES * 416);        // N*100 f32
  float* out = (float*)d_out;

  zero_kernel<<<2048, 256, 0, stream>>>(h_sum, (size_t)NNODES * 101);

  edge_kernel<<<NEDGES / (EW * 16), EW * 32, EDGE_LDS, stream>>>(
      lu, ei, t, msg, tw, tb, featW, featb, tlng, tlnb, tf0W, tf0b, tf1W, tf1b,
      h_sum, cnt);

  const int nodeBlocks = (NNODES + EW * 16 - 1) / (EW * 16);
  node_ffn0_kernel<<<nodeBlocks, EW * 32, K2A_LDS, stream>>>(
      h_sum, cnt, clng, clnb, cf0W, cf0b, hc1);
  node_ffn1_kernel<<<nodeBlocks, EW * 32, K2B_LDS, stream>>>(
      h_sum, cnt, hc1, cf1W, cf1b, projW, projb, out_pre);
  node_out_kernel<<<nodeBlocks, EW * 32, K2C_LDS, stream>>>(
      x, out_pre, resW, resb, olng, olnb, out);
}
